// AttentionBlock_1331439862271
// MI455X (gfx1250) — matrix operand order repaired and checked by reading, not compiled, not run on hardware
//
#include <hip/hip_runtime.h>
#include <hip/hip_bf16.h>

// ---------------------------------------------------------------------------
// AttentionBlock for MI455X (gfx1250, wave32, WMMA 16x16x32 f16 / f32-accum)
//
// Shapes: B=8, C=512, T=H*W=1024, 64 total heads (B*8), ch=64.
// All GEMMs run on v_wmma_f32_16x16x32_f16; GroupNorm + softmax in fp32.
// Workspace (40 MB, L2-resident on a 192MB-L2 part):
//   gnT  [8][1024][512]  f16   (t-major => contiguous QKV B-fragments)
//   q    [64][1024][64]  f16   (A-matrix layout for QK^T, pre-scaled)
//   k    [64][1024][64]  f16   (B-matrix layout for QK^T, pre-scaled)
//   v    [64][64][1024]  f16   (B-matrix layout for PV)
//   a    [8][1024][512]  f16   (A-matrix layout for projection GEMM)
// ---------------------------------------------------------------------------

typedef __attribute__((ext_vector_type(16))) _Float16 v16h;
typedef __attribute__((ext_vector_type(8)))  float    v8f;

#define B_   8
#define C_   512
#define T_   1024
#define CH_  64
#define QKSCALE 0.35355339059f   /* 64^(-1/4) */

__device__ __forceinline__ v8f wmma16x16x32(v16h a, v16h b, v8f c) {
  return __builtin_amdgcn_wmma_f32_16x16x32_f16(
      /*neg_a=*/false, a, /*neg_b=*/false, b,
      /*c_mod=*/(short)0, c, /*reuse_a=*/false, /*reuse_b=*/false);
}

// A-fragment (16x32, f16): lanes 0-15 hold rows 0-15 with K 0..15,
// lanes 16-31 same rows with K 16..31.
__device__ __forceinline__ v16h afrag_f16(const _Float16* base, int ld,
                                          int row0, int k0, int lane) {
  const int r = lane & 15, half = lane >> 4;
  const _Float16* p = base + (size_t)(row0 + r) * ld + k0 + half * 16;
  v16h a;
#pragma unroll
  for (int i = 0; i < 16; ++i) a[i] = p[i];
  return a;
}

__device__ __forceinline__ v16h afrag_f32(const float* base, int ld,
                                          int row0, int k0, int lane) {
  const int r = lane & 15, half = lane >> 4;
  const float* p = base + (size_t)(row0 + r) * ld + k0 + half * 16;
  v16h a;
#pragma unroll
  for (int i = 0; i < 16; ++i) a[i] = (_Float16)p[i];
  return a;
}

// B-fragment (32x16, f16): lanes 0-15 hold cols 0-15 with K 0..15,
// lanes 16-31 with K 16..31; VGPR j packs K=2j,2j+1.  Source is stored
// column-contiguous in K:  elem(k,n) = baseT[(col0+n)*ld + k0 + k].
__device__ __forceinline__ v16h bfrag_f16(const _Float16* baseT, int ld,
                                          int col0, int k0, int lane) {
  const int n = lane & 15, half = lane >> 4;
  const _Float16* p = baseT + (size_t)(col0 + n) * ld + k0 + half * 16;
  return *(const v16h*)p;   // 32B contiguous, 32B aligned at all call sites
}

__device__ __forceinline__ v16h bfrag_f32(const float* baseT, int ld,
                                          int col0, int k0, int lane) {
  const int n = lane & 15, half = lane >> 4;
  const float* p = baseT + (size_t)(col0 + n) * ld + k0 + half * 16;
  v16h b;
#pragma unroll
  for (int i = 0; i < 16; ++i) b[i] = (_Float16)p[i];
  return b;
}

// ---------------------------------------------------------------------------
// Kernel 1: GroupNorm32 (fp32), output f16 transposed [b][t][c]
// One block per (b, group): 16 channels x 1024 positions = 16384 elements.
// ---------------------------------------------------------------------------
__global__ void gn_kernel(const float* __restrict__ x,
                          const float* __restrict__ sc,
                          const float* __restrict__ bi,
                          _Float16* __restrict__ gnT) {
  const int b = blockIdx.x >> 5, g = blockIdx.x & 31;
  const int tid = threadIdx.x;
  const float* xb = x + ((size_t)(b * C_ + g * 16) << 10);

  float s = 0.f, s2 = 0.f;
  for (int i = tid; i < 16 * T_; i += 256) {
    float vv = xb[i];
    s += vv; s2 += vv * vv;
  }
  __shared__ float r1[256], r2[256];
  __shared__ float mean_s, rstd_s;
  r1[tid] = s; r2[tid] = s2;
  __syncthreads();
  for (int off = 128; off > 0; off >>= 1) {
    if (tid < off) { r1[tid] += r1[tid + off]; r2[tid] += r2[tid + off]; }
    __syncthreads();
  }
  if (tid == 0) {
    float mean = r1[0] * (1.f / 16384.f);
    float var  = r2[0] * (1.f / 16384.f) - mean * mean;
    mean_s = mean;
    rstd_s = rsqrtf(var + 1e-5f);
  }
  __syncthreads();
  const float mean = mean_s, rstd = rstd_s;
  for (int i = tid; i < 16 * T_; i += 256) {
    int cc = i >> 10, t = i & 1023;
    int c = g * 16 + cc;
    float vg = (xb[i] - mean) * rstd * sc[c] + bi[c];
    gnT[((((size_t)b) << 10) + t) * C_ + c] = (_Float16)vg;
  }
}

// ---------------------------------------------------------------------------
// Kernel 2: QKV GEMM  qkv[o,t] = qkv_w[o,:] . gn[:,t] + qkv_b[o]
// M=o (1536), N=t (1024), K=c (512). Each wave: 16x64 output strip.
// Scatter-stores into head-major q/k/v layouts; q,k pre-scaled by 64^-1/4.
// ---------------------------------------------------------------------------
__global__ void qkv_kernel(const float* __restrict__ w,
                           const float* __restrict__ bias,
                           const _Float16* __restrict__ gnT,
                           _Float16* __restrict__ q,
                           _Float16* __restrict__ k,
                           _Float16* __restrict__ v) {
  const int lane = threadIdx.x & 31;
  const int wv = blockIdx.x * 8 + (threadIdx.x >> 5);
  const int b = wv / 1536, rem = wv % 1536;
  const int o0 = (rem >> 4) * 16;      // 96 o-tiles
  const int t0 = (rem & 15) * 64;      // 16 t-strips of 64

  const _Float16* gb = gnT + ((((size_t)b) << 10)) * C_;
  v8f acc[4] = {};
  for (int c0 = 0; c0 < C_; c0 += 32) {
    v16h af = afrag_f32(w, C_, o0, c0, lane);
#pragma unroll
    for (int j = 0; j < 4; ++j) {
      v16h bf = bfrag_f16(gb, C_, t0 + j * 16, c0, lane);
      acc[j] = wmma16x16x32(af, bf, acc[j]);
    }
  }
  const int half = lane >> 4, n = lane & 15;
#pragma unroll
  for (int j = 0; j < 4; ++j) {
#pragma unroll
    for (int r = 0; r < 8; ++r) {
      const int o = o0 + r + 8 * half;           // row (output channel)
      const int t = t0 + j * 16 + n;             // col (position)
      float val = acc[j][r] + bias[o];
      const int head = b * 8 + o / 192;          // legacy head layout
      const int part = (o % 192) / 64;           // 0=q 1=k 2=v
      const int cc = o & 63;
      if (part == 0)
        q[((((size_t)head) << 10) + t) * CH_ + cc] = (_Float16)(val * QKSCALE);
      else if (part == 1)
        k[((((size_t)head) << 10) + t) * CH_ + cc] = (_Float16)(val * QKSCALE);
      else
        v[(((size_t)head) * CH_ + cc) * T_ + t] = (_Float16)val;
    }
  }
}

// ---------------------------------------------------------------------------
// Kernel 3: attention. One block per (head, 16-row query tile).
// Phase 1: scores S[16][1024] -> LDS (QK^T via WMMA, K=64).
// Phase 2: fp32 softmax in LDS (exp in place, 1/rowsum kept for PV epilogue).
// Phase 3: PV via WMMA, split-K across wave pairs, LDS combine,
//          store a[b][t][c] f16 (A-layout for the projection GEMM).
// Dynamic LDS: 64KB scores + ~5KB scratch.
// ---------------------------------------------------------------------------
__global__ void attn_kernel(const _Float16* __restrict__ q,
                            const _Float16* __restrict__ k,
                            const _Float16* __restrict__ v,
                            _Float16* __restrict__ abuf) {
  const int head = blockIdx.x >> 6, tblk = blockIdx.x & 63;
  const int lane = threadIdx.x & 31, wv = threadIdx.x >> 5;
  const int half = lane >> 4, n = lane & 15;

  extern __shared__ char smem[];
  float* S    = (float*)smem;                 // [16][1024]
  float* mrow = (float*)(smem + 65536);       // [16]
  float* invl = mrow + 16;                    // [16]
  float* red  = invl + 16;                    // [16][16]
  float* pd   = red + 256;                    // [4][16][16] split-K partials

  const _Float16* qh = q + (((size_t)head) << 10) * CH_;
  const _Float16* kh = k + (((size_t)head) << 10) * CH_;
  const _Float16* vh = v + ((size_t)head) * CH_ * T_;
  const int t0 = tblk * 16;

  // ---- Phase 1: scores (each wave: 8 s-tiles of 16) ----
  v16h aq0 = afrag_f16(qh, CH_, t0, 0,  lane);
  v16h aq1 = afrag_f16(qh, CH_, t0, 32, lane);
  for (int j = 0; j < 8; ++j) {
    const int s0 = (wv * 8 + j) * 16;
    v16h b0 = bfrag_f16(kh, CH_, s0, 0,  lane);
    v16h b1 = bfrag_f16(kh, CH_, s0, 32, lane);
    v8f accs = {};
    accs = wmma16x16x32(aq0, b0, accs);
    accs = wmma16x16x32(aq1, b1, accs);
#pragma unroll
    for (int r = 0; r < 8; ++r)
      S[(r + 8 * half) * T_ + s0 + n] = accs[r];
  }
  __syncthreads();

  // ---- Phase 2: fp32 softmax, 16 threads per row ----
  {
    const int row = threadIdx.x >> 4, sub = threadIdx.x & 15;
    float* Sr = S + row * T_ + sub * 64;
    float mx = -3.4e38f;
    for (int i = 0; i < 64; ++i) mx = fmaxf(mx, Sr[i]);
    red[row * 16 + sub] = mx;
    __syncthreads();
    if (sub == 0) {
      float m = red[row * 16];
      for (int i = 1; i < 16; ++i) m = fmaxf(m, red[row * 16 + i]);
      mrow[row] = m;
    }
    __syncthreads();
    const float m = mrow[row];
    float s = 0.f;
    for (int i = 0; i < 64; ++i) {
      float e = __expf(Sr[i] - m);
      Sr[i] = e;                       // exp in place; scale folded in later
      s += e;
    }
    red[row * 16 + sub] = s;
    __syncthreads();
    if (sub == 0) {
      float tsum = 0.f;
      for (int i = 0; i < 16; ++i) tsum += red[row * 16 + i];
      invl[row] = 1.0f / tsum;
    }
    __syncthreads();
  }

  // ---- Phase 3: PV, split-K over wave pairs (ctile = wv&3, shalf = wv>>2) ---
  const int ctile = wv & 3, shalf = wv >> 2;
  const int c0 = ctile * 16;
  v8f acc = {};
  for (int s0 = shalf * 512; s0 < shalf * 512 + 512; s0 += 32) {
    // A-fragment of exp'd probabilities straight from LDS (f32 -> f16)
    const float* Sp = S + (size_t)(lane & 15) * T_ + s0 + half * 16;
    v16h ap;
#pragma unroll
    for (int i = 0; i < 16; ++i) ap[i] = (_Float16)Sp[i];
    v16h bv = bfrag_f16(vh, T_, c0, s0, lane);
    acc = wmma16x16x32(ap, bv, acc);
  }
  if (shalf == 1) {
#pragma unroll
    for (int r = 0; r < 8; ++r)
      pd[ctile * 256 + (r + 8 * half) * 16 + n] = acc[r];
  }
  __syncthreads();
  if (shalf == 0) {
    const int b = head >> 3, hin = head & 7;
#pragma unroll
    for (int r = 0; r < 8; ++r) {
      const int row = r + 8 * half;
      float val = (acc[r] + pd[ctile * 256 + row * 16 + n]) * invl[row];
      const int t = t0 + row;
      const int c = hin * CH_ + c0 + n;
      abuf[((((size_t)b) << 10) + t) * C_ + c] = (_Float16)val;
    }
  }
}

// ---------------------------------------------------------------------------
// Kernel 4: projection + bias + residual.
// out[b,o,t] = x[b,o,t] + proj_b[o] + sum_c proj_w[o,c] * a[b,t,c]
// M=t, N=o, K=c; each wave: 16x64 strip.
// ---------------------------------------------------------------------------
__global__ void proj_kernel(const _Float16* __restrict__ abuf,
                            const float* __restrict__ pw,
                            const float* __restrict__ pb,
                            const float* __restrict__ x,
                            float* __restrict__ out) {
  const int lane = threadIdx.x & 31;
  const int wv = blockIdx.x * 8 + (threadIdx.x >> 5);
  const int b = wv / 512, rem = wv % 512;
  const int t0 = (rem >> 3) * 16;   // 64 t-tiles
  const int o0 = (rem & 7) * 64;    // 8 o-strips

  const _Float16* ab = abuf + ((((size_t)b) << 10)) * C_;
  v8f acc[4] = {};
  for (int c0 = 0; c0 < C_; c0 += 32) {
    v16h af = afrag_f16(ab, C_, t0, c0, lane);
#pragma unroll
    for (int j = 0; j < 4; ++j) {
      v16h bf = bfrag_f32(pw, C_, o0 + j * 16, c0, lane);
      acc[j] = wmma16x16x32(af, bf, acc[j]);
    }
  }
  const int half = lane >> 4, n = lane & 15;
#pragma unroll
  for (int j = 0; j < 4; ++j) {
#pragma unroll
    for (int r = 0; r < 8; ++r) {
      const int o = o0 + j * 16 + n;
      const int t = t0 + r + 8 * half;
      const size_t idx = (((size_t)(b * C_ + o)) << 10) + t;
      out[idx] = x[idx] + pb[o] + acc[j][r];
    }
  }
}

// ---------------------------------------------------------------------------
extern "C" void kernel_launch(void* const* d_in, const int* in_sizes, int n_in,
                              void* d_out, int out_size, void* d_ws, size_t ws_size,
                              hipStream_t stream) {
  const float* x     = (const float*)d_in[0];
  const float* nsc   = (const float*)d_in[1];
  const float* nbi   = (const float*)d_in[2];
  const float* qkv_w = (const float*)d_in[3];
  const float* qkv_b = (const float*)d_in[4];
  const float* pw    = (const float*)d_in[5];
  const float* pb    = (const float*)d_in[6];
  float* out = (float*)d_out;

  // Workspace carve-up (f16 intermediates, 40 MB total, L2-resident).
  char* ws = (char*)d_ws;
  _Float16* gnT = (_Float16*)(ws);                    //  8 MB
  _Float16* q   = (_Float16*)(ws + (8u  << 20));      //  8 MB
  _Float16* k   = (_Float16*)(ws + (16u << 20));      //  8 MB
  _Float16* v   = (_Float16*)(ws + (24u << 20));      //  8 MB
  _Float16* a   = (_Float16*)(ws + (32u << 20));      //  8 MB

  const int attn_lds = 65536 + 64 + 64 + 1024 + 4096; // 70784 B
  (void)hipFuncSetAttribute((const void*)attn_kernel,
                            hipFuncAttributeMaxDynamicSharedMemorySize,
                            attn_lds);

  // 1) GroupNorm -> gnT[b][t][c] f16
  gn_kernel<<<B_ * 32, 256, 0, stream>>>(x, nsc, nbi, gnT);
  // 2) QKV GEMM -> q/k/v head-major f16 (q,k pre-scaled)
  qkv_kernel<<<(B_ * 96 * 16) / 8, 256, 0, stream>>>(qkv_w, qkv_b, gnT, q, k, v);
  // 3) attention -> a[b][t][c] f16
  attn_kernel<<<64 * 64, 256, attn_lds, stream>>>(q, k, v, a);
  // 4) projection + residual -> out f32
  proj_kernel<<<(B_ * 64 * 8) / 8, 256, 0, stream>>>(a, pw, pb, x, out);
}
